// DeformableStripConv_45758581571670
// MI455X (gfx1250) — matrix operand order repaired and checked by reading, not compiled, not run on hardware
//
#include <hip/hip_runtime.h>

#define Bn 2
#define Cn 64
#define Hn 256
#define Wn 256
#define On 64
#define Kn 9
#define CK (Cn*Kn)      // 576
#define HW (Hn*Wn)      // 65536
#define TP 32           // pixels per deform tile
#define SROW 584        // padded S_t row stride in halfs (bank-conflict-free)

typedef __attribute__((ext_vector_type(16))) _Float16 v16h;
typedef __attribute__((ext_vector_type(8)))  _Float16 v8h;
typedef __attribute__((ext_vector_type(8)))  float    v8f;

// ---------------------------------------------------------------------------
// Kernel 1: offset conv  (1x9, C=64 -> 18 channels), fp32 VALU.
// One block per (b,h) row; x row staged in LDS with halo; weights scalarized.
// ---------------------------------------------------------------------------
__global__ void offset_conv_kernel(const float* __restrict__ x,
                                   const float* __restrict__ ow,
                                   const float* __restrict__ ob,
                                   float* __restrict__ offs) {
  __shared__ float tile[Wn + 8];
  const int bh = blockIdx.x;          // 0 .. B*H-1
  const int b  = bh / Hn;
  const int h  = bh % Hn;
  const int t  = threadIdx.x;         // 0..255 == output w

  float acc[18];
  #pragma unroll
  for (int oc = 0; oc < 18; ++oc) acc[oc] = ob[oc];

  for (int c = 0; c < Cn; ++c) {
    __syncthreads();
    const float* xrow = x + ((size_t)(b*Cn + c)*Hn + h)*Wn;
    for (int i = t; i < Wn + 8; i += 256) {
      int xw = i - 4;
      tile[i] = (xw >= 0 && xw < Wn) ? xrow[xw] : 0.0f;
    }
    __syncthreads();
    float xv[9];
    #pragma unroll
    for (int j = 0; j < 9; ++j) xv[j] = tile[t + j];
    #pragma unroll
    for (int oc = 0; oc < 18; ++oc) {
      const float* wrow = ow + (oc*Cn + c)*9;   // lane-uniform -> scalar loads
      #pragma unroll
      for (int j = 0; j < 9; ++j)
        acc[oc] = fmaf(xv[j], wrow[j], acc[oc]);
    }
  }
  #pragma unroll
  for (int oc = 0; oc < 18; ++oc)
    offs[((size_t)(b*18 + oc)*Hn + h)*Wn + t] = acc[oc];
}

// ---------------------------------------------------------------------------
// Kernel 2: pack deform_w into per-lane A fragments for V_WMMA_F32_16X16X32_F16.
// ISA 16-bit A 16x32 layout: lane l -> M = l&15, half = l>>4;
//   VGPR v<4  holds K = half*8 + 2v + {0,1}
//   VGPR v>=4 holds K = 16 + half*8 + 2(v-4) + {0,1}
// Flat layout: wp[(((mt*18 + ks)*32 + lane)*16) + i]  (32B contiguous per lane)
// ---------------------------------------------------------------------------
__global__ void pack_w_kernel(const float* __restrict__ dw,
                              _Float16* __restrict__ wp) {
  int t = blockIdx.x*blockDim.x + threadIdx.x;     // 0 .. 4*18*32-1
  if (t >= 4*18*32) return;
  int lane = t & 31;
  int ks   = (t >> 5) % 18;
  int mt   = t / (18*32);
  int m  = mt*16 + (lane & 15);
  int hh = lane >> 4;
  _Float16* dst = wp + (size_t)t * 16;
  #pragma unroll
  for (int i = 0; i < 16; ++i) {
    int v = i >> 1;
    int k = 32*ks + hh*8 + ((v < 4) ? (2*v) : (16 + 2*(v-4))) + (i & 1);
    dst[i] = (_Float16)dw[m*CK + k];               // dw flat: o*576 + c*9 + kw
  }
}

// ---------------------------------------------------------------------------
// Kernel 3: deformable sampling + WMMA GEMM.
// Per block: 32 consecutive pixels in one image row.
//   phase 1: bilinear params (clamped corner indices + masked weights) -> LDS
//   phase 2: gather/blend S[r=c*9+k][p], store transposed f16 S_t[p][r] (padded)
//   phase 3: 8 waves = 4 M-tiles x 2 N-tiles, 18 steps of wmma_f32_16x16x32_f16
// ---------------------------------------------------------------------------
__global__ void deform_gemm_kernel(const float* __restrict__ x,
                                   const float* __restrict__ offs,
                                   const _Float16* __restrict__ wp,
                                   const float* __restrict__ dbias,
                                   float* __restrict__ out) {
  extern __shared__ char smem[];
  _Float16* St   = (_Float16*)smem;                          // [TP][SROW] halfs
  int*      pidx = (int*)(smem + TP*SROW*2);                 // [9][TP][4]
  float*    pwgt = (float*)(smem + TP*SROW*2 + 9*TP*4*4);    // [9][TP][4]

  const int t   = threadIdx.x;
  const int g0  = blockIdx.x * TP;        // global pixel base (b*HW + h*W + w0)
  const int b   = g0 / HW;
  const int hw0 = g0 % HW;
  const int h   = hw0 / Wn;
  const int w0  = hw0 % Wn;

  // ---- phase 1: bilinear parameters per (k, p) ----
  for (int e = t; e < 9*TP; e += 256) {
    int k = e / TP;
    int p = e % TP;
    int w = w0 + p;
    float dy = offs[((size_t)(b*18 + 2*k    )*Hn + h)*Wn + w];
    float dx = offs[((size_t)(b*18 + 2*k + 1)*Hn + h)*Wn + w];
    float py = dy + (float)h;               // ky == 0 (KH=1, PH=0)
    float px = dx + (float)(w + k - 4);     // kx = k - 4
    float y0f = floorf(py), x0f = floorf(px);
    float ly = py - y0f, lx = px - x0f;
    int y0 = (int)y0f, x0 = (int)x0f;
    float wts[4] = { (1.f-ly)*(1.f-lx), (1.f-ly)*lx, ly*(1.f-lx), ly*lx };
    const int dyv[4] = {0,0,1,1}, dxv[4] = {0,1,0,1};
    #pragma unroll
    for (int j = 0; j < 4; ++j) {
      int yi = y0 + dyv[j], xi = x0 + dxv[j];
      bool valid = (yi >= 0) && (yi < Hn) && (xi >= 0) && (xi < Wn);
      int yc = min(max(yi, 0), Hn-1);
      int xc = min(max(xi, 0), Wn-1);
      pidx[(k*TP + p)*4 + j] = yc*Wn + xc;
      pwgt[(k*TP + p)*4 + j] = valid ? wts[j] : 0.0f;
    }
  }
  __syncthreads();

  // ---- phase 2: gather + blend, store S transposed as f16 ----
  {
    const int p = t & 31;
    for (int r = (t >> 5); r < CK; r += 8) {
      int c = r / 9;
      int k = r - c*9;
      const float* xb = x + (size_t)(b*Cn + c)*HW;
      int j0 = (k*TP + p)*4;
      float s = pwgt[j0+0]*xb[pidx[j0+0]]
              + pwgt[j0+1]*xb[pidx[j0+1]]
              + pwgt[j0+2]*xb[pidx[j0+2]]
              + pwgt[j0+3]*xb[pidx[j0+3]];
      St[p*SROW + r] = (_Float16)s;
    }
  }
  __syncthreads();

  // ---- phase 3: WMMA GEMM  D[64 x 32] = W[64 x 576] * S[576 x 32] ----
  const int wave = t >> 5;
  const int lane = t & 31;
  const int mt  = wave & 3;      // M tile (16 out channels)
  const int nt  = wave >> 2;     // N tile (16 pixels)
  const int l15 = lane & 15;
  const int hh  = lane >> 4;

  v8f acc = {};
  const _Float16* wlane = wp + ((size_t)(mt*18)*32 + lane)*16;
  const _Float16* srow  = St + (nt*16 + l15)*SROW + hh*8;

  for (int ks = 0; ks < 18; ++ks) {
    v8h a0 = *(const v8h*)(wlane + (size_t)ks*32*16);
    v8h a1 = *(const v8h*)(wlane + (size_t)ks*32*16 + 8);
    v8h b0 = *(const v8h*)(srow + ks*32);
    v8h b1 = *(const v8h*)(srow + ks*32 + 16);
    v16h a, bf;
    #pragma unroll
    for (int i = 0; i < 8; ++i) {
      a[i] = a0[i]; a[8+i] = a1[i];
      bf[i] = b0[i]; bf[8+i] = b1[i];
    }
    acc = __builtin_amdgcn_wmma_f32_16x16x32_f16(
        /*neg_a=*/false, a, /*neg_b=*/false, bf,
        /*c_mod=*/(short)0, acc, /*reuse_a=*/false, /*reuse_b=*/false);
  }

  // ---- store D + bias (C/D layout: VGPR v -> M = v + hh*8) ----
  const int pix = hw0 + nt*16 + l15;
  #pragma unroll
  for (int v = 0; v < 8; ++v) {
    int o = mt*16 + v + hh*8;
    out[(size_t)(b*On + o)*HW + pix] = acc[v] + dbias[o];
  }
}

// ---------------------------------------------------------------------------
extern "C" void kernel_launch(void* const* d_in, const int* in_sizes, int n_in,
                              void* d_out, int out_size, void* d_ws, size_t ws_size,
                              hipStream_t stream) {
  const float* x  = (const float*)d_in[0];   // (2,64,256,256)
  const float* ow = (const float*)d_in[1];   // (18,64,1,9)
  const float* ob = (const float*)d_in[2];   // (18,)
  const float* dw = (const float*)d_in[3];   // (64,64,1,9)
  const float* db = (const float*)d_in[4];   // (64,)
  float* out = (float*)d_out;                // (2,64,256,256)

  const size_t offs_bytes = (size_t)Bn * 18 * HW * sizeof(float);  // 9.4 MB
  float*    offs = (float*)d_ws;
  _Float16* wpk  = (_Float16*)((char*)d_ws + offs_bytes);          // 73.7 KB

  // 1) offsets = conv1x9(x) + bias
  offset_conv_kernel<<<Bn*Hn, 256, 0, stream>>>(x, ow, ob, offs);

  // 2) pack deform weights into WMMA A-fragment order
  pack_w_kernel<<<(4*18*32 + 255)/256, 256, 0, stream>>>(dw, wpk);

  // 3) sample + GEMM
  const size_t lds = (size_t)TP*SROW*2 + (size_t)9*TP*4*4 * 2;     // 46592 B
  deform_gemm_kernel<<<(Bn*HW)/TP, 256, lds, stream>>>(x, offs, wpk, db, out);
}
